// CL_3839700763191
// MI455X (gfx1250) — matrix operand order repaired and checked by reading, not compiled, not run on hardware
//
#include <hip/hip_runtime.h>
#include <hip/hip_bf16.h>
#include <math.h>

#define BNUM 8192
#define DDIM 1024
#define KNEG 30

typedef __attribute__((ext_vector_type(16))) _Float16 v16h;
typedef __attribute__((ext_vector_type(8)))  _Float16 v8h;
typedef __attribute__((ext_vector_type(4)))  _Float16 v4h;
typedef __attribute__((ext_vector_type(8)))  float    v8f;

// ---------------------------------------------------------------------------
// Kernel 1: L2-normalize each row of z_i / z_j and emit an f16 pool [2B][D].
// One wave32 per row: each lane owns 8 float4 groups (8*4*32 = 1024 elems).
// ---------------------------------------------------------------------------
__global__ __launch_bounds__(256) void nrm_pool_kernel(
    const float* __restrict__ zi, const float* __restrict__ zj,
    _Float16* __restrict__ pool) {
  const int wid  = threadIdx.x >> 5;
  const int lane = threadIdx.x & 31;
  const int row  = blockIdx.x * 8 + wid;            // 0 .. 2B-1
  const float* src = (row < BNUM) ? (zi + (size_t)row * DDIM)
                                  : (zj + (size_t)(row - BNUM) * DDIM);

  float4 v[8];
  float ss = 0.0f;
#pragma unroll
  for (int g = 0; g < 8; ++g) {
    v[g] = *(const float4*)(src + (size_t)(g * 32 + lane) * 4);
    ss += v[g].x * v[g].x + v[g].y * v[g].y + v[g].z * v[g].z + v[g].w * v[g].w;
  }
#pragma unroll
  for (int m = 16; m; m >>= 1) ss += __shfl_xor(ss, m, 32);

  const float n   = sqrtf(ss);
  const float inv = 1.0f / fmaxf(n, 1e-12f);        // matches F.normalize semantics

  _Float16* dst = pool + (size_t)row * DDIM;
#pragma unroll
  for (int g = 0; g < 8; ++g) {
    v4h h;
    h.x = (_Float16)(v[g].x * inv);
    h.y = (_Float16)(v[g].y * inv);
    h.z = (_Float16)(v[g].z * inv);
    h.w = (_Float16)(v[g].w * inv);
    *(v4h*)(dst + (size_t)(g * 32 + lane) * 4) = h;
  }
}

// ---------------------------------------------------------------------------
// Kernel 2: per-row similarities via v_wmma_f32_16x16x32_f16 + log-softmax.
// One wave per row iteration. A = 2 tiles of 16 gathered target rows,
// B = zi[b] K-chunk broadcast across all 16 columns (columns redundant).
// Fragment layouts per CDNA5 ISA 7.12.2.
// ---------------------------------------------------------------------------
__global__ __launch_bounds__(256) void simloss_kernel(
    const int* __restrict__ negidx, const float* __restrict__ temp_p,
    const _Float16* __restrict__ pool, float* __restrict__ row_loss) {
  __shared__ float sims[8][32];

  const int wid  = threadIdx.x >> 5;
  const int lane = threadIdx.x & 31;
  const int gw   = blockIdx.x * 8 + wid;            // 0 .. 2047
  const float invT = 1.0f / temp_p[0];

  const int hi   = lane >> 4;                       // 0: lanes 0-15, 1: lanes 16-31
  const int aoff = hi * 8;                          // A-frag K sub-offset
  const int boff = hi * 16;                         // unused; B loads use aoff
  const int tl   = lane & 15;                       // target slot within tile

  for (int it = 0; it < 4; ++it) {                  // 2048 waves * 4 rows = 8192
    const int b = gw + it * 2048;

    // Target row indices: slot 0 -> positive (zj row = b+B), 1..30 -> negatives,
    // slot 31 -> pad (dup of positive, excluded from softmax).
    int t0 = tl;                                    // tile 0: targets 0..15
    int t1 = tl + 16;                               // tile 1: targets 16..31
    int r0 = (t0 == 0) ? (b + BNUM) : negidx[b * KNEG + (t0 - 1)];
    int r1 = (t1 <= KNEG) ? negidx[b * KNEG + (t1 - 1)] : (b + BNUM);

    const _Float16* rowA0 = pool + (size_t)r0 * DDIM;
    const _Float16* rowA1 = pool + (size_t)r1 * DDIM;
    const _Float16* rowB  = pool + (size_t)b  * DDIM;

    v8f c0 = {}; v8f c1 = {};
#pragma unroll 4
    for (int kb = 0; kb < DDIM; kb += 32) {
      union { v16h v; v8h h[2]; } a0, a1, bb;
      // A 16x32 f16: lanes<16 hold K {kb..kb+7},{kb+16..kb+23}; lanes>=16 shift by 8
      a0.h[0] = *(const v8h*)(rowA0 + kb + aoff);
      a0.h[1] = *(const v8h*)(rowA0 + kb + 16 + aoff);
      a1.h[0] = *(const v8h*)(rowA1 + kb + aoff);
      a1.h[1] = *(const v8h*)(rowA1 + kb + 16 + aoff);
      // B 32x16 f16, all columns identical: same K layout as A (lanes<16 {kb..kb+7},{kb+16..kb+23}; lanes>=16 shift by 8)
      bb.h[0] = *(const v8h*)(rowB + kb + aoff);
      bb.h[1] = *(const v8h*)(rowB + kb + 16 + aoff);

      c0 = __builtin_amdgcn_wmma_f32_16x16x32_f16(false, a0.v, false, bb.v,
                                                  (short)0, c0, false, false);
      c1 = __builtin_amdgcn_wmma_f32_16x16x32_f16(false, a1.v, false, bb.v,
                                                  (short)0, c1, false, false);
    }

    // D layout: VGPR r -> M=r (lanes 0-15) / M=8+r (lanes 16-31); columns identical.
    if (lane == 0) {
#pragma unroll
      for (int r = 0; r < 8; ++r) { sims[wid][r]      = c0[r]; sims[wid][16 + r] = c1[r]; }
    } else if (lane == 16) {
#pragma unroll
      for (int r = 0; r < 8; ++r) { sims[wid][8 + r]  = c0[r]; sims[wid][24 + r] = c1[r]; }
    }
    __syncthreads();                                // uniform trip count: safe

    // 31-way log-softmax across the wave; lane t owns logit t (t=0 is positive).
    const float x     = sims[wid][lane] * invT;
    const bool  valid = (lane < 31);
    float xm = valid ? x : -__builtin_inff();
#pragma unroll
    for (int m = 16; m; m >>= 1) xm = fmaxf(xm, __shfl_xor(xm, m, 32));
    float e = valid ? expf(x - xm) : 0.0f;
#pragma unroll
    for (int m = 16; m; m >>= 1) e += __shfl_xor(e, m, 32);
    if (lane == 0) row_loss[b] = -(x - xm - logf(e));
    __syncthreads();
  }
}

// ---------------------------------------------------------------------------
// Kernel 3: deterministic tree reduction of per-row losses -> mean.
// ---------------------------------------------------------------------------
__global__ __launch_bounds__(256) void reduce_kernel(
    const float* __restrict__ row_loss, float* __restrict__ out) {
  __shared__ float buf[256];
  float s = 0.0f;
  for (int i = threadIdx.x; i < BNUM; i += 256) s += row_loss[i];
  buf[threadIdx.x] = s;
  __syncthreads();
  for (int off = 128; off; off >>= 1) {
    if (threadIdx.x < off) buf[threadIdx.x] += buf[threadIdx.x + off];
    __syncthreads();
  }
  if (threadIdx.x == 0) out[0] = buf[0] / (float)BNUM;
}

extern "C" void kernel_launch(void* const* d_in, const int* in_sizes, int n_in,
                              void* d_out, int out_size, void* d_ws, size_t ws_size,
                              hipStream_t stream) {
  const float* z_i   = (const float*)d_in[0];
  const float* z_j   = (const float*)d_in[1];
  const float* temp  = (const float*)d_in[2];
  const int*   negix = (const int*)d_in[3];
  float*       out   = (float*)d_out;

  // Workspace: f16 normalized pool [2B][D] (32 MB) + per-row losses (32 KB).
  _Float16* pool     = (_Float16*)d_ws;
  float*    row_loss = (float*)((char*)d_ws + (size_t)2 * BNUM * DDIM * sizeof(_Float16));

  // Pass 1: normalize + build f16 pool (2B rows, 1 wave/row, 8 waves/block).
  nrm_pool_kernel<<<(2 * BNUM) / 8, 256, 0, stream>>>(z_i, z_j, pool);

  // Pass 2: WMMA similarities + log-softmax (2048 waves x 4 rows).
  simloss_kernel<<<256, 256, 0, stream>>>(negix, temp, pool, row_loss);

  // Pass 3: mean reduction.
  reduce_kernel<<<1, 256, 0, stream>>>(row_loss, out);
}
